// FKPNUL_2680059593440
// MI455X (gfx1250) — matrix operand order repaired and checked by reading, not compiled, not run on hardware
//
#include <hip/hip_runtime.h>
#include <hip/hip_bf16.h>
#include <math.h>

typedef __attribute__((ext_vector_type(16))) _Float16 v16h;
typedef __attribute__((ext_vector_type(8)))  _Float16 v8h;
typedef __attribute__((ext_vector_type(8)))  float    v8f;

#define HH 96
#define WW 96
#define NB 4
#define BETA 0.1f
#define NCONV 13

// ---------------------------------------------------------------------------
// Pack all 13 convs' f32 OIHW weights into the CDNA5 WMMA A-matrix (16x32 f16)
// VGPR layout in ONE launch. Layout: [tap][cc][mt][lane 0..31][e 0..15] so the
// conv kernel sees every A address as (A + lane*16) + compile-time constant
// (folded into the 24-bit global_load IOFFSET).
// A-matrix element map: lane = (m%16) + 16*half, K = e + 16*half
// Channels >= Cin (padding up to CinPad, multiple of 32) are zero-filled.
// ---------------------------------------------------------------------------
struct PackDesc {
  const float* src[NCONV];
  _Float16*    dst[NCONV];
  int Cout[NCONV], Cin[NCONV], Pad[NCONV];
  int start[NCONV + 1];
  int total;
};

__global__ __launch_bounds__(256) void pack_all(PackDesc d) {
  int tid = blockIdx.x * blockDim.x + threadIdx.x;
  if (tid >= d.total) return;
  int ci_ = 0;
  while (tid >= d.start[ci_ + 1]) ci_++;
  int t      = tid - d.start[ci_];
  int Cin    = d.Cin[ci_];
  int CinPad = d.Pad[ci_];
  int numC   = CinPad >> 5;
  int numM   = d.Cout[ci_] >> 4;
  int co    = t / (9 * CinPad);
  int r     = t % (9 * CinPad);
  int tap   = r / CinPad;
  int k_all = r % CinPad;
  int cc   = k_all >> 5;
  int k    = k_all & 31;
  int half = k >> 4;
  int e    = k & 15;
  int lane = (co & 15) + (half << 4);
  int mt   = co >> 4;
  float v = (k_all < Cin) ? d.src[ci_][(co * Cin + k_all) * 9 + tap] : 0.0f;
  d.dst[ci_][(((tap * numC + cc) * numM + mt) * 32 + lane) * 16 + e] = (_Float16)v;
}

__global__ __launch_bounds__(256) void zero_f16(_Float16* __restrict__ p, int total) {
  int i = blockIdx.x * blockDim.x + threadIdx.x;
  if (i < total) p[i] = (_Float16)0.0f;
}

// First conv: Cin=1 -> 64, direct (tiny: ~42 MFLOP). Writes channels-last f16
// into both X0 (stride 64) and the RRDB feats buffer F (stride 320, offset 0).
__global__ __launch_bounds__(256) void conv1_direct(
    const float* __restrict__ depth, const float* __restrict__ Wc,
    const float* __restrict__ bias, _Float16* __restrict__ X0,
    _Float16* __restrict__ F) {
  int tid = blockIdx.x * blockDim.x + threadIdx.x;
  if (tid >= NB * HH * WW * 64) return;
  int co  = tid & 63;
  int pos = tid >> 6;
  int w = pos % WW; int h = (pos / WW) % HH; int n = pos / (WW * HH);
  float s = bias[co];
#pragma unroll
  for (int kh = 0; kh < 3; kh++) {
    int hi = h + kh - 1;
    if (hi < 0 || hi >= HH) continue;
#pragma unroll
    for (int kw = 0; kw < 3; kw++) {
      int wi = w + kw - 1;
      if (wi < 0 || wi >= WW) continue;
      s += Wc[co * 9 + kh * 3 + kw] * depth[(n * HH + hi) * WW + wi];
    }
  }
  X0[(size_t)pos * 64 + co] = (_Float16)s;
  F [(size_t)pos * 320 + co] = (_Float16)s;
}

// ---------------------------------------------------------------------------
// 3x3 SAME conv as implicit GEMM with V_WMMA_F32_16X16X32_F16.
// One wave computes a 16*NUMM (Cout) x 16 (contiguous w) output tile.
// Fully unrolled over taps (9), K-chunks (NUMC) and Cout tiles (NUMM); the
// input channel stride SIN is a template constant so all B row addressing is
// shift/add (no v_mul_u64).
//   B (32x16): lane holds spatial column n = w0 + lane%16; K = e + 16*(lane/16)
//              -> one aligned v16h load per chunk, cc*64B immediate offsets.
//   A (16x32): prepacked per-lane v16h; all offsets compile-time constants.
// Epilogue (all fused): bias, optional ReLU, optional residual chain
//   v = acc+bias; if Res:  v = Res + beta*v; if Res2: v = Res2 + beta*v
// and contiguous-channel v8h (b128) stores to Y and optionally Y2.
// ---------------------------------------------------------------------------
template <int NUMM, int NUMC, int SIN>
__global__ __launch_bounds__(128) void conv3x3_wmma(
    const _Float16* __restrict__ X,
    const _Float16* __restrict__ A, const float* __restrict__ bias,
    _Float16* __restrict__ Y, int SOut, int Obase, int relu,
    const _Float16* __restrict__ Res, int SRes,
    const _Float16* __restrict__ Res2, int SRes2, float beta,
    _Float16* __restrict__ Y2, int SOut2) {
  const int WT = WW / 16;                       // 6 w-tiles per row
  int wave = threadIdx.x >> 5;
  int lane = threadIdx.x & 31;
  int tile = blockIdx.x * 4 + wave;             // NB*HH*WT tiles total
  if (tile >= NB * HH * WT) return;
  int wt_ = tile % WT;
  int h   = (tile / WT) % HH;
  int n   = tile / (WT * HH);
  int w0  = wt_ * 16;
  int lm   = lane & 15;
  int half = lane >> 4;
  int wcol = w0 + lm;

  const _Float16* Abase = A + lane * 16;        // all A offsets are immediates

  v8f acc[NUMM];
#pragma unroll
  for (int m = 0; m < NUMM; m++) acc[m] = {};

#pragma unroll
  for (int kh = 0; kh < 3; kh++) {
    int hi = h + kh - 1;
    bool hv = (hi >= 0) && (hi < HH);
#pragma unroll
    for (int kw = 0; kw < 3; kw++) {
      int wi = wcol + kw - 1;
      bool valid = hv && (wi >= 0) && (wi < WW);
      const _Float16* bp = nullptr;
      if (valid)
        bp = X + (size_t)((n * HH + hi) * WW + wi) * SIN + half * 16;
      const int tap = kh * 3 + kw;
#pragma unroll
      for (int cc = 0; cc < NUMC; cc++) {
        v16h b = {};
        if (valid) b = *(const v16h*)(bp + cc * 32);          // 32B aligned
#pragma unroll
        for (int m = 0; m < NUMM; m++) {
          v16h a = *(const v16h*)(Abase + ((tap * NUMC + cc) * NUMM + m) * 512);
          acc[m] = __builtin_amdgcn_wmma_f32_16x16x32_f16(
              false, a, false, b, (short)0, acc[m], false, false);
        }
      }
    }
  }

  // C/D layout: vgpr r, lane -> (M = r + 8*half, N = lane%16).
  // Channel group co = m*16 + half*8 + r is contiguous -> b128 stores.
  size_t pos = (size_t)((n * HH + h) * WW + wcol);
#pragma unroll
  for (int m = 0; m < NUMM; m++) {
    int co0 = m * 16 + half * 8;
    float vv[8];
#pragma unroll
    for (int r = 0; r < 8; r++) {
      float v = acc[m][r] + bias[co0 + r];
      if (relu) v = v > 0.0f ? v : 0.0f;
      vv[r] = v;
    }
    if (Res) {
      v8h rv = *(const v8h*)(Res + pos * SRes + co0);
#pragma unroll
      for (int r = 0; r < 8; r++) vv[r] = (float)rv[r] + beta * vv[r];
    }
    if (Res2) {
      v8h rv = *(const v8h*)(Res2 + pos * SRes2 + co0);
#pragma unroll
      for (int r = 0; r < 8; r++) vv[r] = (float)rv[r] + beta * vv[r];
    }
    v8h o;
#pragma unroll
    for (int r = 0; r < 8; r++) o[r] = (_Float16)vv[r];
    *(v8h*)(Y + pos * SOut + Obase + co0) = o;
    if (Y2) *(v8h*)(Y2 + pos * SOut2 + co0) = o;
  }
}

// ---------------------------------------------------------------------------
// Fused epilogue at 768x768: outer-product kernel weights (24x24), pixel
// shuffle r=8, sigmoid*2-1, 3x3 unfold of depth nearest-up'd, weighted sum,
// plus nearest-up residual. One 64-thread group per low-res pixel; the 48
// KPN outputs and 9 depth taps are staged in LDS.
// ---------------------------------------------------------------------------
__global__ __launch_bounds__(256) void fuse_out(
    const _Float16* __restrict__ WT, const float* __restrict__ depth,
    float* __restrict__ out) {
  __shared__ float s_wt[4][48];
  __shared__ float s_tap[4][9];
  int g = threadIdx.x >> 6;
  int t = threadIdx.x & 63;
  int pix = blockIdx.x * 4 + g;                 // over NB*HH*WW
  if (pix >= NB * HH * WW) return;
  int w = pix % WW; int h = (pix / WW) % HH; int n = pix / (WW * HH);
  if (t < 48) {
    s_wt[g][t] = (float)WT[(size_t)pix * 64 + t];
  } else if (t < 57) {
    int tap = t - 48; int ki = tap / 3, kj = tap % 3;
    int hi = h + ki - 1, wi = w + kj - 1;
    float dv = (hi >= 0 && hi < HH && wi >= 0 && wi < WW)
                   ? depth[(n * HH + hi) * WW + wi] : 0.0f;
    s_tap[g][tap] = dv;
  }
  __syncthreads();
  int dy = t >> 3, dx = t & 7;
  float sum = 0.0f;
#pragma unroll
  for (int tp = 0; tp < 9; tp++) {
    int idx = tp * 64 + dy * 8 + dx;            // pixel-shuffle inverse
    int a = idx / 24, c = idx % 24;             // einsum channel = a*24 + c
    float wv = s_wt[g][a] * s_wt[g][24 + c];
    wv = 2.0f / (1.0f + __expf(-wv)) - 1.0f;    // sigmoid*2 - 1
    sum += s_tap[g][tp] * wv;
  }
  sum += s_tap[g][4];                           // nearest_up(depth) residual
  int Ho = h * 8 + dy, Wo = w * 8 + dx;
  out[((size_t)n * (HH * 8) + Ho) * (WW * 8) + Wo] = sum;
}

// ---------------------------------------------------------------------------
extern "C" void kernel_launch(void* const* d_in, const int* in_sizes, int n_in,
                              void* d_out, int out_size, void* d_ws, size_t ws_size,
                              hipStream_t stream) {
  (void)in_sizes; (void)n_in; (void)out_size; (void)ws_size;
  const float* depth  = (const float*)d_in[1];
  const float* conv_w = (const float*)d_in[2];
  const float* conv_b = (const float*)d_in[3];
  const float *b1_ws[5], *b1_bs[5], *b2_ws[5], *b2_bs[5];
  for (int i = 0; i < 5; i++) {
    b1_ws[i] = (const float*)d_in[4 + i];
    b1_bs[i] = (const float*)d_in[9 + i];
    b2_ws[i] = (const float*)d_in[14 + i];
    b2_bs[i] = (const float*)d_in[19 + i];
  }
  const float *kpn_ws[3], *kpn_bs[3];
  for (int i = 0; i < 3; i++) {
    kpn_ws[i] = (const float*)d_in[24 + i];
    kpn_bs[i] = (const float*)d_in[27 + i];
  }
  float* out = (float*)d_out;

  // ---- workspace carve (all f16 activation buffers channels-last) ----
  const size_t PIX = (size_t)NB * HH * WW;      // 36864
  char* p = (char*)d_ws;
  auto carve = [&](size_t bytes) {
    char* r = p; p += (bytes + 255) & ~(size_t)255; return r;
  };
  _Float16* F   = (_Float16*)carve(PIX * 320 * 2);  // RRDB dense-concat feats
  _Float16* X0  = (_Float16*)carve(PIX * 64 * 2);
  _Float16* X1  = (_Float16*)carve(PIX * 64 * 2);
  _Float16* WT0 = (_Float16*)carve(PIX * 64 * 2);
  _Float16* WK1 = (_Float16*)carve(PIX * 64 * 2);   // WK1/WK2 contiguous
  _Float16* WK2 = (_Float16*)carve(PIX * 64 * 2);
  _Float16* WTF = (_Float16*)carve(PIX * 64 * 2);

  // ---- packed-weight descriptor (13 convs in one pack launch) ----
  PackDesc pd;
  for (int i = 0; i < 5; i++) {
    pd.Cout[i] = 64;     pd.Cin[i] = 64 * (i + 1);     pd.Pad[i] = 64 * (i + 1);
    pd.Cout[5 + i] = 64; pd.Cin[5 + i] = 64 * (i + 1); pd.Pad[5 + i] = 64 * (i + 1);
    pd.src[i] = b1_ws[i]; pd.src[5 + i] = b2_ws[i];
  }
  pd.Cout[10] = 48; pd.Cin[10] = 64; pd.Pad[10] = 64; pd.src[10] = kpn_ws[0];
  pd.Cout[11] = 48; pd.Cin[11] = 48; pd.Pad[11] = 64; pd.src[11] = kpn_ws[1];
  pd.Cout[12] = 48; pd.Cin[12] = 48; pd.Pad[12] = 64; pd.src[12] = kpn_ws[2];
  int acc = 0;
  _Float16* PW[NCONV];
  for (int i = 0; i < NCONV; i++) {
    pd.start[i] = acc;
    int cnt = pd.Cout[i] * 9 * pd.Pad[i];
    PW[i] = (_Float16*)carve((size_t)cnt * 2);
    pd.dst[i] = PW[i];
    acc += cnt;
  }
  pd.start[NCONV] = acc;
  pd.total = acc;

  int totE = (int)(PIX * 64);

  // ---- zero pad channels of the 48-wide KPN buffers (poison-safe) ----
  zero_f16<<<(2 * totE + 255) / 256, 256, 0, stream>>>(WK1, 2 * totE);

  // ---- pack all weights (one launch) ----
  pack_all<<<(pd.total + 255) / 256, 256, 0, stream>>>(pd);

  // ---- conv1 (1 -> 64), writes X0 and feats[0:64) ----
  conv1_direct<<<(totE + 255) / 256, 256, 0, stream>>>(depth, conv_w, conv_b, X0, F);

  const int tiles = NB * HH * (WW / 16);        // 2304
  dim3 cg((tiles + 3) / 4), cb(128);
  const _Float16* NR = (const _Float16*)nullptr;
  _Float16* NY = (_Float16*)nullptr;

  // ---- RRDB 1 (dense convs write concat channels in place) ----
  conv3x3_wmma<4, 2,320><<<cg, cb, 0, stream>>>(F, PW[0], b1_bs[0], F, 320,  64, 1, NR,0, NR,0, 0.0f, NY,0);
  conv3x3_wmma<4, 4,320><<<cg, cb, 0, stream>>>(F, PW[1], b1_bs[1], F, 320, 128, 1, NR,0, NR,0, 0.0f, NY,0);
  conv3x3_wmma<4, 6,320><<<cg, cb, 0, stream>>>(F, PW[2], b1_bs[2], F, 320, 192, 1, NR,0, NR,0, 0.0f, NY,0);
  conv3x3_wmma<4, 8,320><<<cg, cb, 0, stream>>>(F, PW[3], b1_bs[3], F, 320, 256, 1, NR,0, NR,0, 0.0f, NY,0);
  // conv5: X1 = X0 + B*(acc+b); dual-store X1 and feats[0:64) for RRDB2
  conv3x3_wmma<4,10,320><<<cg, cb, 0, stream>>>(F, PW[4], b1_bs[4], X1, 64, 0, 0, X0,64, NR,0, BETA, F,320);

  // ---- RRDB 2 ----
  conv3x3_wmma<4, 2,320><<<cg, cb, 0, stream>>>(F, PW[5], b2_bs[0], F, 320,  64, 1, NR,0, NR,0, 0.0f, NY,0);
  conv3x3_wmma<4, 4,320><<<cg, cb, 0, stream>>>(F, PW[6], b2_bs[1], F, 320, 128, 1, NR,0, NR,0, 0.0f, NY,0);
  conv3x3_wmma<4, 6,320><<<cg, cb, 0, stream>>>(F, PW[7], b2_bs[2], F, 320, 192, 1, NR,0, NR,0, 0.0f, NY,0);
  conv3x3_wmma<4, 8,320><<<cg, cb, 0, stream>>>(F, PW[8], b2_bs[3], F, 320, 256, 1, NR,0, NR,0, 0.0f, NY,0);
  // conv5 + final residual chain fused:
  //   WT0 = X0 + B*( X1 + B*(acc+bias) )   (== x0 + BETA*rrdb2_out)
  conv3x3_wmma<4,10,320><<<cg, cb, 0, stream>>>(F, PW[9], b2_bs[4], WT0, 64, 0, 0, X1,64, X0,64, BETA, NY,0);

  // ---- KPN head (48 outputs in stride-64 buffers, pad zeroed) ----
  conv3x3_wmma<3, 2, 64><<<cg, cb, 0, stream>>>(WT0, PW[10], kpn_bs[0], WK1, 64, 0, 1, NR,0, NR,0, 0.0f, NY,0);
  conv3x3_wmma<3, 2, 64><<<cg, cb, 0, stream>>>(WK1, PW[11], kpn_bs[1], WK2, 64, 0, 1, NR,0, NR,0, 0.0f, NY,0);
  conv3x3_wmma<3, 2, 64><<<cg, cb, 0, stream>>>(WK2, PW[12], kpn_bs[2], WTF, 64, 0, 0, NR,0, NR,0, 0.0f, NY,0);

  // ---- fused outer-product / pixel-shuffle / sigmoid / unfold epilogue ----
  fuse_out<<<(int)(PIX / 4), 256, 0, stream>>>(WTF, depth, out);
}
